// Model_84146999263755
// MI455X (gfx1250) — matrix operand order repaired and checked by reading, not compiled, not run on hardware
//
#include <hip/hip_runtime.h>
#include <hip/hip_bf16.h>
#include <math.h>

typedef __attribute__((ext_vector_type(16))) __bf16 v16bf;
typedef __attribute__((ext_vector_type(8)))  float  v8f;
typedef unsigned int u32;
typedef __attribute__((ext_vector_type(4))) u32 u32x4;
typedef __attribute__((ext_vector_type(8))) int i32x8;
typedef __attribute__((ext_vector_type(4))) int i32x4;

constexpr int B_   = 4;
constexpr int C_   = 256;
constexpr int S_   = 4096;
constexpr int DFF_ = 1024;
constexpr int NB_  = 128;
constexpr int G_   = 1;

#define DEVFN __device__ __forceinline__

DEVFN v8f wmma_bf16(v16bf a, v16bf b, v8f c) {
  return __builtin_amdgcn_wmma_f32_16x16x32_bf16(false, a, false, b, (short)0, c,
                                                 false, false);
}

// A-matrix 16x32 (MxK) bf16 fragment from row-major f32 memory.
// Lanes 0-15 -> M=lane, K=0..15; lanes 16-31 -> M=lane-16, K=16..31.
DEVFN v16bf loadA_f32(const float* base, int ld, int lane) {
  int m  = lane & 15;
  int kb = (lane & 16) ? 16 : 0;
  const float* p = base + (size_t)m * ld + kb;
  v16bf a;
#pragma unroll
  for (int h = 0; h < 8; ++h) a[h] = (__bf16)p[h];
#pragma unroll
  for (int h = 0; h < 8; ++h) a[8 + h] = (__bf16)p[8 + h];
  return a;
}

DEVFN v16bf loadA_bf16(const __bf16* base, int ld, int lane) {
  int m  = lane & 15;
  int kb = (lane & 16) ? 16 : 0;
  const __bf16* p = base + (size_t)m * ld + kb;
  v16bf a;
#pragma unroll
  for (int h = 0; h < 8; ++h) a[h] = p[h];
#pragma unroll
  for (int h = 0; h < 8; ++h) a[8 + h] = p[8 + h];
  return a;
}

// B-matrix 32x16 (KxN), element (k,n) at base[k*ld + n].
// Lanes 0-15: N=lane, K=0..15; lanes 16-31: K=16..31.
DEVFN v16bf loadB_f32(const float* base, int ld, int lane) {
  int n  = lane & 15;
  int kb = (lane & 16) ? 16 : 0;
  const float* p = base + (size_t)kb * ld + n;
  v16bf b;
#pragma unroll
  for (int h = 0; h < 16; ++h) b[h] = (__bf16)p[(size_t)h * ld];
  return b;
}

// B-matrix 32x16 (KxN) where memory holds the transpose: (k,n) at base[n*ld + k].
DEVFN v16bf loadBt_f32(const float* base, int ld, int lane) {
  int n  = lane & 15;
  int kb = (lane & 16) ? 16 : 0;
  const float* p = base + (size_t)n * ld + kb;
  v16bf b;
#pragma unroll
  for (int h = 0; h < 16; ++h) b[h] = (__bf16)p[h];
  return b;
}

DEVFN float wave_sum(float v) {
#pragma unroll
  for (int o = 16; o; o >>= 1) v += __shfl_xor(v, o, 32);
  return v;
}
DEVFN float wave_max(float v) {
#pragma unroll
  for (int o = 16; o; o >>= 1) v = fmaxf(v, __shfl_xor(v, o, 32));
  return v;
}

// ---------------------------------------------------------------------------
// Tensor Data Mover: DMA a 2D fp32 tile (tile_w x tile_h, row stride
// row_stride elems) from global into LDS at lds_byte_addr. ISA ch.8 D#.
// Issue from ONE wave only (EXEC ignored; instruction is per-wave).
// ---------------------------------------------------------------------------
DEVFN void tdm_load_2d_f32(u32 lds_byte_addr, const float* gaddr,
                           u32 tile_w, u32 tile_h,
                           u32 tensor_w, u32 tensor_h, u32 row_stride) {
  unsigned long long ga = (unsigned long long)(uintptr_t)gaddr;
  u32x4 g0;
  g0[0] = 1u;                                           // count=1, user mode
  g0[1] = lds_byte_addr;                                // lds_addr
  g0[2] = (u32)ga;                                      // global_addr[31:0]
  g0[3] = (u32)((ga >> 32) & 0x1FFFFFFu) | (2u << 30);  // addr[56:32] | type=2
  i32x8 g1;
  g1[0] = 0x20000;                                     // data_size=2 (4 bytes)
  g1[1] = (int)(tensor_w << 16);                       // tensor_dim0[15:0]
  g1[2] = (int)((tensor_w >> 16) | (tensor_h << 16));  // td0[31:16] | td1[15:0]
  g1[3] = (int)((tensor_h >> 16) | (tile_w << 16));    // td1[31:16] | tile_dim0
  g1[4] = (int)(tile_h & 0xFFFFu);                     // tile_dim1 (tile_dim2=0)
  g1[5] = (int)row_stride;                             // tensor_dim0_stride[31:0]
  g1[6] = 0;
  g1[7] = 0;
  i32x4 z4 = {0, 0, 0, 0};
#if defined(__clang_major__) && __clang_major__ >= 23
  i32x8 z8 = {0, 0, 0, 0, 0, 0, 0, 0};
  __builtin_amdgcn_tensor_load_to_lds(g0, g1, z4, z4, z8, 0);
#else
  __builtin_amdgcn_tensor_load_to_lds(g0, g1, z4, z4, 0);
#endif
}

// ---------------------------------------------------------------------------
// Kernel 1: x = LN(w_proj @ src + b_proj)  for src1 (z=0) and src2 (z=1).
// ---------------------------------------------------------------------------
__global__ void proj_ln_kernel(const float* __restrict__ src1,
                               const float* __restrict__ src2,
                               const float* __restrict__ wproj,
                               const float* __restrict__ bproj,
                               const float* __restrict__ g,
                               const float* __restrict__ be,
                               float* __restrict__ x1, float* __restrict__ x2) {
  __shared__ float tile[C_][33];
  const int s0 = blockIdx.x * 32;
  const int b  = blockIdx.y;
  const float* src = blockIdx.z ? src2 : src1;
  float* dst       = blockIdx.z ? x2 : x1;
  const int wv = threadIdx.x >> 5, lane = threadIdx.x & 31;
  const int m0 = wv * 32;

  v8f acc[2][2] = {};
  for (int k0 = 0; k0 < C_; k0 += 32) {
    v16bf a0 = loadA_f32(wproj + (size_t)m0 * C_ + k0, C_, lane);
    v16bf a1 = loadA_f32(wproj + (size_t)(m0 + 16) * C_ + k0, C_, lane);
    const float* sb = src + (size_t)b * C_ * S_ + (size_t)k0 * S_ + s0;
    v16bf b0 = loadB_f32(sb, S_, lane);
    v16bf b1 = loadB_f32(sb + 16, S_, lane);
    acc[0][0] = wmma_bf16(a0, b0, acc[0][0]);
    acc[0][1] = wmma_bf16(a0, b1, acc[0][1]);
    acc[1][0] = wmma_bf16(a1, b0, acc[1][0]);
    acc[1][1] = wmma_bf16(a1, b1, acc[1][1]);
  }
  const int mlo = (lane & 16) ? 8 : 0;
  const int nn  = lane & 15;
#pragma unroll
  for (int mt = 0; mt < 2; ++mt)
#pragma unroll
    for (int nt = 0; nt < 2; ++nt)
#pragma unroll
      for (int r = 0; r < 8; ++r) {
        int o = m0 + mt * 16 + mlo + r;
        tile[o][nt * 16 + nn] = acc[mt][nt][r] + bproj[o];
      }
  __syncthreads();

  for (int jj = 0; jj < 4; ++jj) {
    int j = wv * 4 + jj;
    float s = 0.f;
#pragma unroll
    for (int t = 0; t < 8; ++t) s += tile[lane * 8 + t][j];
    float mean = wave_sum(s) * (1.0f / C_);
    float v = 0.f;
#pragma unroll
    for (int t = 0; t < 8; ++t) {
      float d = tile[lane * 8 + t][j] - mean;
      v += d * d;
    }
    float inv = rsqrtf(wave_sum(v) * (1.0f / C_) + 1e-5f);
    float* dp = dst + ((size_t)b * S_ + s0 + j) * C_;
#pragma unroll
    for (int t = 0; t < 8; ++t) {
      int o = lane * 8 + t;
      dp[o] = (tile[o][j] - mean) * inv * g[o] + be[o];
    }
  }
}

// ---------------------------------------------------------------------------
// Kernel 2: global-block attention, flash-style. One block per batch.
// Wave w owns key/dim tiles nt = {2w, 2w+1}, both query M-tiles.
// ---------------------------------------------------------------------------
__global__ void attn_global_kernel(const float* __restrict__ x1,
                                   const float* __restrict__ x2,
                                   const float* __restrict__ g,
                                   const float* __restrict__ be,
                                   float* __restrict__ hout) {
  extern __shared__ float sm[];
  float* Ssc  = sm;               // 32*256 scores / P chunk
  float* Oac  = sm + 32 * 256;    // 32*256 output accumulator
  float* mrow = Oac + 32 * 256;   // 32
  float* lrow = mrow + 32;        // 32
  float* frow = lrow + 32;        // 32
  const int b    = blockIdx.x;
  const int wv   = threadIdx.x >> 5, lane = threadIdx.x & 31;
  const int tid  = threadIdx.x;
  const int mlo  = (lane & 16) ? 8 : 0;
  const int nn   = lane & 15;
  const int nt0  = wv * 2;

  for (int i = tid; i < 32 * 256; i += 256) Oac[i] = 0.f;
  if (tid < 32) { mrow[tid] = -3.0e38f; lrow[tid] = 0.f; frow[tid] = 1.f; }
  __syncthreads();

  const float* qbase  = x1 + (size_t)b * S_ * C_;
  const float* kvbase = x2 + (size_t)b * S_ * C_;

  for (int c = 0; c < S_ / 256; ++c) {
    const int kc0 = c * 256;
    // ---- scores: 2x2 register blocking, shared fragments ----
    {
      v8f acc[2][2] = {};
      for (int d0 = 0; d0 < C_; d0 += 32) {
        v16bf a0 = loadA_f32(qbase + d0, C_, lane);
        v16bf a1 = loadA_f32(qbase + (size_t)16 * C_ + d0, C_, lane);
        v16bf b0 = loadBt_f32(kvbase + (size_t)(kc0 + nt0 * 16) * C_ + d0, C_, lane);
        v16bf b1 = loadBt_f32(kvbase + (size_t)(kc0 + nt0 * 16 + 16) * C_ + d0, C_, lane);
        acc[0][0] = wmma_bf16(a0, b0, acc[0][0]);
        acc[0][1] = wmma_bf16(a0, b1, acc[0][1]);
        acc[1][0] = wmma_bf16(a1, b0, acc[1][0]);
        acc[1][1] = wmma_bf16(a1, b1, acc[1][1]);
      }
#pragma unroll
      for (int mt = 0; mt < 2; ++mt)
#pragma unroll
        for (int nt = 0; nt < 2; ++nt)
#pragma unroll
          for (int r = 0; r < 8; ++r)
            Ssc[(mt * 16 + mlo + r) * 256 + (nt0 + nt) * 16 + nn] =
                acc[mt][nt][r] * 0.0625f;
    }
    __syncthreads();
    // ---- online softmax update (wave handles 4 rows) ----
    for (int rr = 0; rr < 4; ++rr) {
      int r = wv * 4 + rr;
      float mx = -3.0e38f;
#pragma unroll
      for (int jj = 0; jj < 8; ++jj) mx = fmaxf(mx, Ssc[r * 256 + lane + jj * 32]);
      mx = wave_max(mx);
      float mo = mrow[r];
      float mn = fmaxf(mo, mx);
      float f  = __expf(mo - mn);
      float s = 0.f;
#pragma unroll
      for (int jj = 0; jj < 8; ++jj) {
        float p = __expf(Ssc[r * 256 + lane + jj * 32] - mn);
        Ssc[r * 256 + lane + jj * 32] = p;
        s += p;
      }
      s = wave_sum(s);
      if (lane == 0) { mrow[r] = mn; lrow[r] = lrow[r] * f + s; frow[r] = f; }
    }
    __syncthreads();
    // ---- O = O*f + P @ V : 2x2 blocking ----
    {
      v8f acc[2][2] = {};
      for (int k0 = 0; k0 < 256; k0 += 32) {
        v16bf a0 = loadA_f32(Ssc + k0, 256, lane);
        v16bf a1 = loadA_f32(Ssc + (size_t)16 * 256 + k0, 256, lane);
        v16bf b0 = loadB_f32(kvbase + (size_t)(kc0 + k0) * C_ + nt0 * 16, C_, lane);
        v16bf b1 = loadB_f32(kvbase + (size_t)(kc0 + k0) * C_ + nt0 * 16 + 16, C_, lane);
        acc[0][0] = wmma_bf16(a0, b0, acc[0][0]);
        acc[0][1] = wmma_bf16(a0, b1, acc[0][1]);
        acc[1][0] = wmma_bf16(a1, b0, acc[1][0]);
        acc[1][1] = wmma_bf16(a1, b1, acc[1][1]);
      }
#pragma unroll
      for (int mt = 0; mt < 2; ++mt)
#pragma unroll
        for (int nt = 0; nt < 2; ++nt)
#pragma unroll
          for (int r = 0; r < 8; ++r) {
            int m = mt * 16 + mlo + r;
            int n = (nt0 + nt) * 16 + nn;
            Oac[m * 256 + n] = Oac[m * 256 + n] * frow[m] + acc[mt][nt][r];
          }
    }
    __syncthreads();
  }
  // ---- finalize: h = LN(x1 + O/l) ----
  for (int rr = 0; rr < 4; ++rr) {
    int r = wv * 4 + rr;
    float linv = 1.0f / lrow[r];
    float s = 0.f;
#pragma unroll
    for (int jj = 0; jj < 8; ++jj) {
      int cix = lane + jj * 32;
      float xv = qbase[(size_t)r * C_ + cix] + Oac[r * 256 + cix] * linv;
      Ssc[r * 256 + cix] = xv;
      s += xv;
    }
    float mean = wave_sum(s) * (1.0f / C_);
    float v = 0.f;
#pragma unroll
    for (int jj = 0; jj < 8; ++jj) {
      float d = Ssc[r * 256 + lane + jj * 32] - mean;
      v += d * d;
    }
    float inv = rsqrtf(wave_sum(v) * (1.0f / C_) + 1e-5f);
    float* dp = hout + ((size_t)b * S_ + r) * C_;
#pragma unroll
    for (int jj = 0; jj < 8; ++jj) {
      int cix = lane + jj * 32;
      dp[cix] = (Ssc[r * 256 + cix] - mean) * inv * g[cix] + be[cix];
    }
  }
}

// ---------------------------------------------------------------------------
// Kernel 3: sparse BigBird rows. One block per (query block 1..127, batch).
// ---------------------------------------------------------------------------
__global__ void attn_sparse_kernel(const float* __restrict__ x1,
                                   const float* __restrict__ x2,
                                   const int* __restrict__ bidx,
                                   const unsigned char* __restrict__ bval,
                                   int K,
                                   const float* __restrict__ g,
                                   const float* __restrict__ be,
                                   float* __restrict__ hout) {
  extern __shared__ float sm[];
  float* Pr = sm;                       // 32 x (K*32) probabilities
  float* Oc = sm + (size_t)32 * K * 32; // 32 x 256 output
  const int row = blockIdx.x;           // 0..NB-2
  const int b   = blockIdx.y;
  const int qs0 = (row + 1) * 32;
  const int ldp = K * 32;
  const int wv = threadIdx.x >> 5, lane = threadIdx.x & 31;
  const int mlo = (lane & 16) ? 8 : 0;
  const int nn  = lane & 15;
  const float* qbase  = x1 + ((size_t)b * S_ + qs0) * C_;
  const float* kvbase = x2 + (size_t)b * S_ * C_;

  // ---- scores: per 16-key half tile nt, both M tiles share the B fragment ----
  for (int nt = wv; nt < 2 * K; nt += 8) {
    int kb = nt >> 1, half = nt & 1;
    int kidx  = bidx[row * K + kb];
    int valid = bval[row * K + kb];
    if (!valid) {
#pragma unroll
      for (int mt = 0; mt < 2; ++mt)
#pragma unroll
        for (int r = 0; r < 8; ++r)
          Pr[(mt * 16 + mlo + r) * ldp + nt * 16 + nn] = -1.0e9f;
    } else {
      v8f acc0 = {}, acc1 = {};
      const float* kp = kvbase + ((size_t)kidx * 32 + half * 16) * C_;
      for (int d0 = 0; d0 < C_; d0 += 32) {
        v16bf a0 = loadA_f32(qbase + d0, C_, lane);
        v16bf a1 = loadA_f32(qbase + (size_t)16 * C_ + d0, C_, lane);
        v16bf bb = loadBt_f32(kp + d0, C_, lane);
        acc0 = wmma_bf16(a0, bb, acc0);
        acc1 = wmma_bf16(a1, bb, acc1);
      }
#pragma unroll
      for (int r = 0; r < 8; ++r) {
        Pr[(mlo + r) * ldp + nt * 16 + nn]      = acc0[r] * 0.0625f;
        Pr[(16 + mlo + r) * ldp + nt * 16 + nn] = acc1[r] * 0.0625f;
      }
    }
  }
  __syncthreads();

  // ---- softmax per row (normalize in place) ----
  for (int rr = 0; rr < 4; ++rr) {
    int r = wv * 4 + rr;
    float mx = -3.0e38f;
    for (int jj = 0; jj < K; ++jj) mx = fmaxf(mx, Pr[r * ldp + lane + jj * 32]);
    mx = wave_max(mx);
    float s = 0.f;
    for (int jj = 0; jj < K; ++jj) {
      float p = __expf(Pr[r * ldp + lane + jj * 32] - mx);
      Pr[r * ldp + lane + jj * 32] = p;
      s += p;
    }
    float inv = 1.0f / wave_sum(s);
    for (int jj = 0; jj < K; ++jj) Pr[r * ldp + lane + jj * 32] *= inv;
  }
  __syncthreads();

  // ---- O = P @ V (gathered), 2x2 blocking: wave owns dim tiles {2w,2w+1} ----
  {
    const int nt0 = wv * 2;
    v8f acc[2][2] = {};
    for (int kb = 0; kb < K; ++kb) {
      int kidx = bidx[row * K + kb];
      const float* vp = kvbase + (size_t)kidx * 32 * C_;
      v16bf a0 = loadA_f32(Pr + kb * 32, ldp, lane);
      v16bf a1 = loadA_f32(Pr + (size_t)16 * ldp + kb * 32, ldp, lane);
      v16bf b0 = loadB_f32(vp + nt0 * 16, C_, lane);
      v16bf b1 = loadB_f32(vp + nt0 * 16 + 16, C_, lane);
      acc[0][0] = wmma_bf16(a0, b0, acc[0][0]);
      acc[0][1] = wmma_bf16(a0, b1, acc[0][1]);
      acc[1][0] = wmma_bf16(a1, b0, acc[1][0]);
      acc[1][1] = wmma_bf16(a1, b1, acc[1][1]);
    }
#pragma unroll
    for (int mt = 0; mt < 2; ++mt)
#pragma unroll
      for (int nt = 0; nt < 2; ++nt)
#pragma unroll
        for (int r = 0; r < 8; ++r)
          Oc[(mt * 16 + mlo + r) * 256 + (nt0 + nt) * 16 + nn] = acc[mt][nt][r];
  }
  __syncthreads();

  // ---- residual + LN(g12) ----
  for (int rr = 0; rr < 4; ++rr) {
    int r = wv * 4 + rr;
    float s = 0.f;
#pragma unroll
    for (int jj = 0; jj < 8; ++jj) {
      int cix = lane + jj * 32;
      float xv = qbase[(size_t)r * C_ + cix] + Oc[r * 256 + cix];
      Oc[r * 256 + cix] = xv;
      s += xv;
    }
    float mean = wave_sum(s) * (1.0f / C_);
    float v = 0.f;
#pragma unroll
    for (int jj = 0; jj < 8; ++jj) {
      float d = Oc[r * 256 + lane + jj * 32] - mean;
      v += d * d;
    }
    float inv = rsqrtf(wave_sum(v) * (1.0f / C_) + 1e-5f);
    float* dp = hout + ((size_t)b * S_ + qs0 + r) * C_;
#pragma unroll
    for (int jj = 0; jj < 8; ++jj) {
      int cix = lane + jj * 32;
      dp[cix] = (Oc[r * 256 + cix] - mean) * inv * g[cix] + be[cix];
    }
  }
}

// ---------------------------------------------------------------------------
// Kernel 4: u = gelu(h @ w1 + b1), bf16 out. TDM stages w1 K-panel in LDS.
// ---------------------------------------------------------------------------
__global__ void ffn1_kernel(const float* __restrict__ h,
                            const float* __restrict__ w1,
                            const float* __restrict__ b1,
                            __bf16* __restrict__ u) {
  __shared__ float Wp[32 * 256];   // 32 K-rows x 256 N-cols of w1 (32 KB)
  const int rr0 = blockIdx.x * 32;
  const int n0  = blockIdx.y * 256;
  const int wv = threadIdx.x >> 5, lane = threadIdx.x & 31;
  const int mlo = (lane & 16) ? 8 : 0;
  const int nn  = lane & 15;
  const int nt0 = wv * 2;

  v8f acc[2][2] = {};
  for (int k0 = 0; k0 < C_; k0 += 32) {
    if (threadIdx.x < 32) {   // wave 0 drives the Tensor Data Mover
      tdm_load_2d_f32((u32)(size_t)(void*)Wp, w1 + (size_t)k0 * DFF_ + n0,
                      /*tile_w=*/256, /*tile_h=*/32,
                      /*tensor_w=*/256, /*tensor_h=*/(u32)(C_ - k0),
                      /*row_stride=*/DFF_);
      __builtin_amdgcn_s_wait_tensorcnt(0);
    }
    __syncthreads();   // panel visible to all waves
    v16bf a0 = loadA_f32(h + (size_t)rr0 * C_ + k0, C_, lane);
    v16bf a1 = loadA_f32(h + (size_t)(rr0 + 16) * C_ + k0, C_, lane);
    v16bf b0 = loadB_f32(Wp + nt0 * 16, 256, lane);
    v16bf b1 = loadB_f32(Wp + nt0 * 16 + 16, 256, lane);
    acc[0][0] = wmma_bf16(a0, b0, acc[0][0]);
    acc[0][1] = wmma_bf16(a0, b1, acc[0][1]);
    acc[1][0] = wmma_bf16(a1, b0, acc[1][0]);
    acc[1][1] = wmma_bf16(a1, b1, acc[1][1]);
    __syncthreads();   // done reading before next TDM overwrite
  }
#pragma unroll
  for (int mt = 0; mt < 2; ++mt)
#pragma unroll
    for (int nt = 0; nt < 2; ++nt)
#pragma unroll
      for (int r = 0; r < 8; ++r) {
        int m = rr0 + mt * 16 + mlo + r;
        int n = n0 + (nt0 + nt) * 16 + nn;
        float y = acc[mt][nt][r] + b1[n];
        y = 0.5f * y * (1.0f + erff(y * 0.70710678118654752f));  // exact gelu
        u[(size_t)m * DFF_ + n] = (__bf16)y;
      }
}

// ---------------------------------------------------------------------------
// Kernel 5: out = transpose(h + u @ w2 + b2). TDM stages w2 K-panel in LDS.
// ---------------------------------------------------------------------------
__global__ void ffn2_kernel(const __bf16* __restrict__ u,
                            const float* __restrict__ w2,
                            const float* __restrict__ b2,
                            const float* __restrict__ h,
                            float* __restrict__ out) {
  __shared__ float Wp[32 * 256];   // 32 K-rows x 256 N-cols of w2 (32 KB)
  const int rr0 = blockIdx.x * 32;
  const int wv = threadIdx.x >> 5, lane = threadIdx.x & 31;
  const int mlo = (lane & 16) ? 8 : 0;
  const int nn  = lane & 15;
  const int nt0 = wv * 2;

  v8f acc[2][2] = {};
  for (int k0 = 0; k0 < DFF_; k0 += 32) {
    if (threadIdx.x < 32) {
      tdm_load_2d_f32((u32)(size_t)(void*)Wp, w2 + (size_t)k0 * C_,
                      /*tile_w=*/256, /*tile_h=*/32,
                      /*tensor_w=*/256, /*tensor_h=*/(u32)(DFF_ - k0),
                      /*row_stride=*/C_);
      __builtin_amdgcn_s_wait_tensorcnt(0);
    }
    __syncthreads();
    v16bf a0 = loadA_bf16(u + (size_t)rr0 * DFF_ + k0, DFF_, lane);
    v16bf a1 = loadA_bf16(u + (size_t)(rr0 + 16) * DFF_ + k0, DFF_, lane);
    v16bf b0 = loadB_f32(Wp + nt0 * 16, 256, lane);
    v16bf b1 = loadB_f32(Wp + nt0 * 16 + 16, 256, lane);
    acc[0][0] = wmma_bf16(a0, b0, acc[0][0]);
    acc[0][1] = wmma_bf16(a0, b1, acc[0][1]);
    acc[1][0] = wmma_bf16(a1, b0, acc[1][0]);
    acc[1][1] = wmma_bf16(a1, b1, acc[1][1]);
    __syncthreads();
  }
#pragma unroll
  for (int mt = 0; mt < 2; ++mt)
#pragma unroll
    for (int nt = 0; nt < 2; ++nt)
#pragma unroll
      for (int r = 0; r < 8; ++r) {
        int m = rr0 + mt * 16 + mlo + r;
        int o = (nt0 + nt) * 16 + nn;
        int bb_ = m / S_, s = m % S_;
        float val = h[(size_t)m * C_ + o] + acc[mt][nt][r] + b2[o];
        out[((size_t)bb_ * C_ + o) * S_ + s] = val;  // (B, C, S)
      }
}

// ---------------------------------------------------------------------------
extern "C" void kernel_launch(void* const* d_in, const int* in_sizes, int n_in,
                              void* d_out, int out_size, void* d_ws, size_t ws_size,
                              hipStream_t stream) {
  const float* src1  = (const float*)d_in[0];
  const float* src2  = (const float*)d_in[1];
  const float* wproj = (const float*)d_in[2];
  const float* bproj = (const float*)d_in[3];
  const float* g13   = (const float*)d_in[4];
  const float* be13  = (const float*)d_in[5];
  const float* g12   = (const float*)d_in[6];
  const float* be12  = (const float*)d_in[7];
  const float* w1    = (const float*)d_in[8];
  const float* b1    = (const float*)d_in[9];
  const float* w2    = (const float*)d_in[10];
  const float* b2    = (const float*)d_in[11];
  const int*   bidx  = (const int*)d_in[12];
  const unsigned char* bval = (const unsigned char*)d_in[13];
  const int K = in_sizes[12] / (NB_ - G_);
  float* out = (float*)d_out;

  const size_t nBSC = (size_t)B_ * S_ * C_;
  float*  x1   = (float*)d_ws;
  float*  x2   = x1 + nBSC;
  float*  hbuf = x2 + nBSC;
  __bf16* u    = (__bf16*)(hbuf + nBSC);

  proj_ln_kernel<<<dim3(S_ / 32, B_, 2), 256, 0, stream>>>(
      src1, src2, wproj, bproj, g13, be13, x1, x2);

  size_t smg = (size_t)(32 * 256 * 2 + 96) * sizeof(float);
  attn_global_kernel<<<dim3(B_), 256, smg, stream>>>(x1, x2, g12, be12, hbuf);

  size_t sms = ((size_t)32 * K * 32 + 32 * 256) * sizeof(float);
  attn_sparse_kernel<<<dim3(NB_ - 1, B_), 256, sms, stream>>>(
      x1, x2, bidx, bval, K, g12, be12, hbuf);

  ffn1_kernel<<<dim3(B_ * S_ / 32, DFF_ / 256), 256, 0, stream>>>(hbuf, w1, b1, u);
  ffn2_kernel<<<dim3(B_ * S_ / 32), 256, 0, stream>>>(u, w2, b2, hbuf, out);
}
